// SelfSupervisedLoss_58437325029511
// MI455X (gfx1250) — compile-verified, matrix operand order checked by reading, not run on hardware
//
#include <hip/hip_runtime.h>
#include <hip/hip_fp16.h>

typedef __attribute__((ext_vector_type(16))) _Float16 v16h;
typedef __attribute__((ext_vector_type(8)))  _Float16 v8h;
typedef __attribute__((ext_vector_type(8)))  float    v8f;

#define N_ROWS   8192
#define N_COLS   256
#define N_TILES  (N_ROWS / 16)   // 512
#define NCLUST   64
#define E16_OFF  65536           // staged f16 buffer offset in d_ws
#define WS_NEEDED (E16_OFF + (size_t)N_ROWS * N_COLS * 2)

// ===========================================================================
// PRIMARY PATH: stage normalized f16 (fragment-swizzled), then pure-f16 WMMA.
// ===========================================================================

// One wave32 per row: butterfly-reduce ||row||^2, scale, convert to f16, and
// store with per-32-K-block swizzle [0:8)|[16:24)|[8:16)|[24:32) so that each
// WMMA lane's 16-half fragment (ISA 7.12.2 16-bit A/B layout) is one
// contiguous 32-byte chunk.
__global__ __launch_bounds__(256) void stage_kernel(const float* __restrict__ emb,
                                                    _Float16* __restrict__ e16,
                                                    float* __restrict__ total) {
  if (blockIdx.x == 0 && threadIdx.x == 0) *total = 0.0f;  // harness doesn't re-zero ws
  const int wave = threadIdx.x >> 5;
  const int lane = threadIdx.x & 31;
  const int row  = blockIdx.x * 8 + wave;                  // grid = N_ROWS/8
  const float* p = emb + (size_t)row * N_COLS + lane * 8;
  const float4 x0 = *(const float4*)(p);
  const float4 x1 = *(const float4*)(p + 4);
  float ss = x0.x * x0.x + x0.y * x0.y + x0.z * x0.z + x0.w * x0.w
           + x1.x * x1.x + x1.y * x1.y + x1.z * x1.z + x1.w * x1.w;
#pragma unroll
  for (int o = 16; o > 0; o >>= 1) ss += __shfl_xor(ss, o, 32);  // all lanes get sum
  const float inv = 1.0f / fmaxf(sqrtf(ss), 1e-12f);
  v8h h;
  h[0] = (_Float16)(x0.x * inv); h[1] = (_Float16)(x0.y * inv);
  h[2] = (_Float16)(x0.z * inv); h[3] = (_Float16)(x0.w * inv);
  h[4] = (_Float16)(x1.x * inv); h[5] = (_Float16)(x1.y * inv);
  h[6] = (_Float16)(x1.z * inv); h[7] = (_Float16)(x1.w * inv);
  const int blk  = lane >> 2;                 // which 32-K block
  const int o8   = (lane & 3) * 8;            // source group within block
  const int dsto = (o8 == 8) ? 16 : (o8 == 16) ? 8 : o8;  // swizzle: swap middle groups
  *(v8h*)(e16 + (size_t)row * N_COLS + blk * 32 + dsto) = h;
}

// Each wave computes a 32x32 Gram tile (2x2 WMMA tiles, 4 accumulators) for
// register-level A/B reuse: 1 global b128 + 1 ds b128 per WMMA instead of 2+2.
// Block = 8 waves sharing one 32-row B tile staged in LDS (16 KB).
__global__ __launch_bounds__(256) void gram32_kernel(const _Float16* __restrict__ e16,
                                                     const long long* __restrict__ labels,
                                                     float* __restrict__ total) {
  __shared__ __align__(32) _Float16 Bs[32 * N_COLS];       // 16 KB: 32 columns (tj2)
  const int wave = threadIdx.x >> 5;
  const int lane = threadIdx.x & 31;
  const int tj2 = blockIdx.x;                              // 32-col group (0..255)
  const int ti2 = blockIdx.y * 8 + wave;                   // 32-row group (0..255)

  { // cooperative copy of B tile (already fragment-swizzled): 1024 x 16B chunks
    const uint4* src = (const uint4*)(e16 + (size_t)tj2 * 32 * N_COLS);
    uint4* dst = (uint4*)Bs;
#pragma unroll
    for (int q = 0; q < 4; ++q)
      dst[threadIdx.x + q * 256] = src[threadIdx.x + q * 256];
  }
  __syncthreads();

  const int r   = lane & 15;
  const int sel = lane >> 4;
  const _Float16* pA0 = e16 + (size_t)(ti2 * 32 + r) * N_COLS;
  const _Float16* pA1 = pA0 + 16 * N_COLS;
  const _Float16* pB0 = Bs + r * N_COLS;
  const _Float16* pB1 = pB0 + 16 * N_COLS;

  v8f c00 = {}, c01 = {}, c10 = {}, c11 = {};
#pragma unroll
  for (int kk = 0; kk < 8; ++kk) {
    const int off = kk * 32 + sel * 16;                    // swizzled fragment offset
    const v16h a0 = *(const v16h*)(pA0 + off);
    const v16h a1 = *(const v16h*)(pA1 + off);
    const v16h b0 = *(const v16h*)(pB0 + off);
    const v16h b1 = *(const v16h*)(pB1 + off);
    c00 = __builtin_amdgcn_wmma_f32_16x16x32_f16(false, a0, false, b0, (short)0, c00, false, false);
    c01 = __builtin_amdgcn_wmma_f32_16x16x32_f16(false, a0, false, b1, (short)0, c01, false, false);
    c10 = __builtin_amdgcn_wmma_f32_16x16x32_f16(false, a1, false, b0, (short)0, c10, false, false);
    c11 = __builtin_amdgcn_wmma_f32_16x16x32_f16(false, a1, false, b1, (short)0, c11, false, false);
  }

  // C layout: N = lane&15, M = v + 8*(lane>>4); dist^2 = 2 - 2*gram (unit rows)
  const int ibase = ti2 * 32, jbase = tj2 * 32;
  const int j0 = jbase + r, j1 = jbase + 16 + r;
  const long long lj0 = labels[j0];
  const long long lj1 = labels[j1];
  float s = 0.0f;
#pragma unroll
  for (int v = 0; v < 8; ++v) {
    const int m  = v + sel * 8;
    const int i0 = ibase + m, i1 = ibase + 16 + m;
    const long long li0 = labels[i0];
    const long long li1 = labels[i1];
    if (li0 == lj0 && i0 != j0) s += sqrtf(fmaxf(2.0f - 2.0f * c00[v], 1e-24f));
    if (li0 == lj1 && i0 != j1) s += sqrtf(fmaxf(2.0f - 2.0f * c01[v], 1e-24f));
    if (li1 == lj0 && i1 != j0) s += sqrtf(fmaxf(2.0f - 2.0f * c10[v], 1e-24f));
    if (li1 == lj1 && i1 != j1) s += sqrtf(fmaxf(2.0f - 2.0f * c11[v], 1e-24f));
  }
#pragma unroll
  for (int o = 16; o > 0; o >>= 1) s += __shfl_down(s, o, 32);
  if (lane == 0) atomicAdd(total, s);
}

// ===========================================================================
// FALLBACK PATH (ws too small to stage): f32 loads + on-the-fly f16 convert.
// ===========================================================================
__global__ __launch_bounds__(256) void norms_kernel(const float* __restrict__ emb,
                                                    float* __restrict__ invn,
                                                    float* __restrict__ total) {
  if (blockIdx.x == 0 && threadIdx.x == 0) *total = 0.0f;
  const int wave = threadIdx.x >> 5;
  const int lane = threadIdx.x & 31;
  const int row  = blockIdx.x * 8 + wave;
  const float* p = emb + (size_t)row * N_COLS + lane * 8;
  const float4 x0 = *(const float4*)(p);
  const float4 x1 = *(const float4*)(p + 4);
  float ss = x0.x * x0.x + x0.y * x0.y + x0.z * x0.z + x0.w * x0.w
           + x1.x * x1.x + x1.y * x1.y + x1.z * x1.z + x1.w * x1.w;
#pragma unroll
  for (int o = 16; o > 0; o >>= 1) ss += __shfl_down(ss, o, 32);
  if (lane == 0) invn[row] = 1.0f / fmaxf(sqrtf(ss), 1e-12f);
}

__device__ __forceinline__ v16h load_frag_f32(const float* __restrict__ p, int k0, float sc) {
  const float4 u0 = *(const float4*)(p + k0);
  const float4 u1 = *(const float4*)(p + k0 + 4);
  const float4 u2 = *(const float4*)(p + k0 + 16);
  const float4 u3 = *(const float4*)(p + k0 + 20);
  v16h f;
  f[0]  = (_Float16)(u0.x * sc); f[1]  = (_Float16)(u0.y * sc);
  f[2]  = (_Float16)(u0.z * sc); f[3]  = (_Float16)(u0.w * sc);
  f[4]  = (_Float16)(u1.x * sc); f[5]  = (_Float16)(u1.y * sc);
  f[6]  = (_Float16)(u1.z * sc); f[7]  = (_Float16)(u1.w * sc);
  f[8]  = (_Float16)(u2.x * sc); f[9]  = (_Float16)(u2.y * sc);
  f[10] = (_Float16)(u2.z * sc); f[11] = (_Float16)(u2.w * sc);
  f[12] = (_Float16)(u3.x * sc); f[13] = (_Float16)(u3.y * sc);
  f[14] = (_Float16)(u3.z * sc); f[15] = (_Float16)(u3.w * sc);
  return f;
}

__global__ __launch_bounds__(256) void gram_f32_kernel(const float* __restrict__ emb,
                                                       const long long* __restrict__ labels,
                                                       const float* __restrict__ invn,
                                                       float* __restrict__ total) {
  const int wave = threadIdx.x >> 5;
  const int lane = threadIdx.x & 31;
  const int ti = blockIdx.y * 8 + wave;
  const int tj = blockIdx.x;
  const int r   = lane & 15;
  const int sel = lane >> 4;
  const int rowA = ti * 16 + r, rowB = tj * 16 + r;
  const float* pA = emb + (size_t)rowA * N_COLS;
  const float* pB = emb + (size_t)rowB * N_COLS;
  const float ia = invn[rowA], ib = invn[rowB];
  v8f c = {};
#pragma unroll
  for (int kk = 0; kk < 8; ++kk) {
    const int k0 = kk * 32 + sel * 8;
    const v16h a = load_frag_f32(pA, k0, ia);
    const v16h b = load_frag_f32(pB, k0, ib);
    c = __builtin_amdgcn_wmma_f32_16x16x32_f16(false, a, false, b,
                                               (short)0, c, false, false);
  }
  const int j = tj * 16 + r;
  const long long lj = labels[j];
  float s = 0.0f;
#pragma unroll
  for (int v = 0; v < 8; ++v) {
    const int i = ti * 16 + v + sel * 8;
    if (labels[i] == lj && i != j)
      s += sqrtf(fmaxf(2.0f - 2.0f * c[v], 1e-24f));
  }
#pragma unroll
  for (int o = 16; o > 0; o >>= 1) s += __shfl_down(s, o, 32);
  if (lane == 0) atomicAdd(total, s);
}

// ===========================================================================
// Finalize: count distinct labels, divide.
// ===========================================================================
__global__ __launch_bounds__(256) void finalize_kernel(const long long* __restrict__ labels,
                                                       const float* __restrict__ total,
                                                       float* __restrict__ out) {
  __shared__ int hist[NCLUST];
  if (threadIdx.x < NCLUST) hist[threadIdx.x] = 0;
  __syncthreads();
  for (int i = threadIdx.x; i < N_ROWS; i += 256)
    atomicAdd(&hist[(int)labels[i]], 1);
  __syncthreads();
  if (threadIdx.x == 0) {
    int c = 0;
    for (int k = 0; k < NCLUST; ++k) c += (hist[k] > 0) ? 1 : 0;
    out[0] = *total / (float)c;
  }
}

extern "C" void kernel_launch(void* const* d_in, const int* in_sizes, int n_in,
                              void* d_out, int out_size, void* d_ws, size_t ws_size,
                              hipStream_t stream) {
  const float*     emb    = (const float*)d_in[0];       // [8192, 256] f32
  const long long* labels = (const long long*)d_in[1];   // [8192] int64
  float* total = (float*)d_ws;

  if (ws_size >= WS_NEEDED) {
    _Float16* e16 = (_Float16*)((char*)d_ws + E16_OFF);
    stage_kernel<<<N_ROWS / 8, 256, 0, stream>>>(emb, e16, total);
    gram32_kernel<<<dim3(N_TILES / 2, N_TILES / 16), 256, 0, stream>>>(e16, labels, total);
  } else {
    float* invn = (float*)((char*)d_ws + 256);
    norms_kernel<<<N_ROWS / 8, 256, 0, stream>>>(emb, invn, total);
    gram_f32_kernel<<<dim3(N_TILES, N_TILES / 8), 256, 0, stream>>>(emb, labels, invn, total);
  }
  finalize_kernel<<<1, 256, 0, stream>>>(labels, total, (float*)d_out);
}
